// UnifiedReasoningGNN_16003048145718
// MI455X (gfx1250) — matrix operand order repaired and checked by reading, not compiled, not run on hardware
//
#include <hip/hip_runtime.h>
#include <hip/hip_bf16.h>

// ---------------------------------------------------------------------------
// UnifiedReasoningGNN forward for gfx1250 (CDNA5, wave32, WMMA)
// All dense matmuls go through v_wmma_f32_16x16x32_f16 (f16 A/B, f32 acc).
// ---------------------------------------------------------------------------

typedef __attribute__((ext_vector_type(16))) _Float16 v16h;
typedef __attribute__((ext_vector_type(8)))  _Float16 v8h;
typedef __attribute__((ext_vector_type(8)))  float    v8f;

#define NN   50000
#define EE   400000
#define HIDC 128
#define BB   64
#define ETOT (EE + NN)

static __device__ __forceinline__ int gtid() {
    return blockIdx.x * blockDim.x + threadIdx.x;
}
static __device__ __forceinline__ int gstride() {
    return gridDim.x * blockDim.x;
}

// monotonic float<->uint key for atomic segment-max
static __device__ __forceinline__ unsigned fkey(float f) {
    unsigned u = __float_as_uint(f);
    return (u & 0x80000000u) ? ~u : (u | 0x80000000u);
}
static __device__ __forceinline__ float unfkey(unsigned k) {
    unsigned u = (k & 0x80000000u) ? (k ^ 0x80000000u) : ~k;
    return __uint_as_float(u);
}
#define NEG_INF_KEY 0x007FFFFFu   // fkey(-inf)

// ------------------------------ utility kernels -----------------------------

__global__ void k_fill_f32(float* p, float v, int n) {
    for (int i = gtid(); i < n; i += gstride()) p[i] = v;
}
__global__ void k_fill_u32(unsigned* p, unsigned v, int n) {
    for (int i = gtid(); i < n; i += gstride()) p[i] = v;
}
__global__ void k_f32_to_f16(const float* __restrict__ s, _Float16* __restrict__ d, int n) {
    for (int i = gtid(); i < n; i += gstride()) d[i] = (_Float16)s[i];
}
// W [K,N] f32 row-major  ->  Bt [Npad,K] f16 (transposed, zero-padded cols)
__global__ void k_wprep(const float* __restrict__ W, _Float16* __restrict__ Bt,
                        int K, int N, int Npad) {
    int total = Npad * K;
    for (int t = gtid(); t < total; t += gstride()) {
        int c = t / K, k = t - c * K;
        Bt[t] = (c < N) ? (_Float16)W[(size_t)k * N + c] : (_Float16)0.0f;
    }
}

// ------------------------------ WMMA GEMM -----------------------------------
// C[M,Nreal] = A[M,K](f16) @ Bt^T  (+bias) (+epilogue); Bt is [Npad,K] f16.
// One wave computes one 16x16 tile. K multiple of 32, Npad multiple of 16.
// ep: 0=none 1=relu 2=sigmoid
__global__ __launch_bounds__(256)
void k_gemm(const _Float16* __restrict__ A, const _Float16* __restrict__ Bt,
            const float* __restrict__ bias, float* __restrict__ C,
            int M, int K, int Npad, int Nreal, int ep) {
    const int lane = threadIdx.x & 31;
    const int wave = blockIdx.x * (blockDim.x >> 5) + (threadIdx.x >> 5);
    const int ntiles = Npad >> 4;
    const int tm = wave / ntiles;
    const int tn = wave - tm * ntiles;
    const int m0 = tm << 4;
    if (m0 >= M) return;                      // wave-uniform exit (EXEC stays all-1s)

    const int half = lane >> 4;               // 0: lanes 0-15, 1: lanes 16-31
    const int l16  = lane & 15;
    int arow = m0 + l16; if (arow >= M) arow = M - 1;   // clamp (rows >= M not stored)

    // A fragment: lane half selects K offset 0/16; 16 contiguous K per lane.
    const _Float16* Ap = A + (size_t)arow * K + (half ? 16 : 0);
    // B fragment: lane half selects K offset 0/16; 16 contiguous K per lane.
    const _Float16* Bp = Bt + (size_t)(tn * 16 + l16) * K + (half ? 16 : 0);

    v8f acc = {};
    for (int k0 = 0; k0 < K; k0 += 32) {
        union { v16h v; struct { v8h lo, hi; } s; } a;
        a.s.lo = *(const v8h*)(Ap + k0);          // K = k0+ko .. +7
        a.s.hi = *(const v8h*)(Ap + k0 + 8);      // K = k0+ko+8 .. +15
        v16h b = *(const v16h*)(Bp + k0);         // K = k0+ko .. +15
        acc = __builtin_amdgcn_wmma_f32_16x16x32_f16(
            false, a.v, false, b, (short)0, acc, false, false);
    }

    const int col = tn * 16 + l16;
    if (col < Nreal) {
        float bv = bias ? bias[col] : 0.0f;
#pragma unroll
        for (int r = 0; r < 8; ++r) {
            int mr = m0 + r + (half ? 8 : 0);
            if (mr < M) {
                float v = acc[r] + bv;
                if (ep == 1)      v = fmaxf(v, 0.0f);
                else if (ep == 2) v = 1.0f / (1.0f + __expf(-v));
                C[(size_t)mr * Nreal + col] = v;
            }
        }
    }
}

// ------------------------- input featurization ------------------------------
// feat16[n,0:64]   = node_emb[clip(int(x[n,0]),0,10)]
// feat16[n,64:128] = pe2(relu(pe1(pos)))
__global__ void k_feat(const float* __restrict__ x, const float* __restrict__ emb,
                       const float* __restrict__ w1, const float* __restrict__ b1,
                       const float* __restrict__ w2, const float* __restrict__ b2,
                       _Float16* __restrict__ feat) {
    for (int n = gtid(); n < NN; n += gstride()) {
        int v = (int)x[n * 3 + 0];
        v = v < 0 ? 0 : (v > 10 ? 10 : v);
        const float* er = emb + v * 64;
        _Float16* f = feat + (size_t)n * HIDC;
        for (int c = 0; c < 64; ++c) f[c] = (_Float16)er[c];
        float px = x[n * 3 + 1], py = x[n * 3 + 2];
        float hid[32];
#pragma unroll
        for (int j = 0; j < 32; ++j) {
            float h = px * w1[j] + py * w1[32 + j] + b1[j];
            hid[j] = fmaxf(h, 0.0f);
        }
        for (int k = 0; k < 64; ++k) {
            float o = b2[k];
#pragma unroll
            for (int j = 0; j < 32; ++j) o += hid[j] * w2[j * 64 + k];
            f[64 + k] = (_Float16)o;
        }
    }
}

// ------------------------------- GAT pieces ---------------------------------

__global__ void k_build_edges(const int* __restrict__ ei, int* __restrict__ s,
                              int* __restrict__ d) {
    for (int t = gtid(); t < ETOT; t += gstride()) {
        if (t < EE) { s[t] = ei[t]; d[t] = ei[EE + t]; }
        else        { s[t] = t - EE; d[t] = t - EE; }
    }
}

// per-(node,head) attention projections: a = sum_c h[n,h,c] * att[h,c]
__global__ void k_att_nodes(const float* __restrict__ h, const float* __restrict__ atS,
                            const float* __restrict__ atD, float* __restrict__ aS,
                            float* __restrict__ aD, int H) {
    int total = NN * H;
    for (int t = gtid(); t < total; t += gstride()) {
        int n = t / H, hd = t - n * H;
        const float* hr = h + (size_t)n * H * HIDC + hd * HIDC;
        const float* s = atS + hd * HIDC;
        const float* d = atD + hd * HIDC;
        float as = 0.f, ad = 0.f;
        for (int c = 0; c < HIDC; ++c) { as += hr[c] * s[c]; ad += hr[c] * d[c]; }
        aS[t] = as; aD[t] = ad;
    }
}

// pass 1: e = leaky_relu(aS[src]+aD[dst]); segment max over dst
__global__ void k_edge_max(const int* __restrict__ s, const int* __restrict__ d,
                           const float* __restrict__ aS, const float* __restrict__ aD,
                           float* __restrict__ eb, unsigned* __restrict__ mk, int H) {
    int total = ETOT * H;
    for (int t = gtid(); t < total; t += gstride()) {
        int e = t / H, hd = t - e * H;
        float v = aS[s[e] * H + hd] + aD[d[e] * H + hd];
        v = v > 0.f ? v : 0.2f * v;
        eb[t] = v;
        atomicMax(&mk[d[e] * H + hd], fkey(v));
    }
}
// pass 2: ex = exp(e - m[dst]); segment sum over dst
__global__ void k_edge_exp(const int* __restrict__ d, float* __restrict__ eb,
                           const unsigned* __restrict__ mk, float* __restrict__ den, int H) {
    int total = ETOT * H;
    for (int t = gtid(); t < total; t += gstride()) {
        int e = t / H, hd = t - e * H;
        float ex = __expf(eb[t] - unfkey(mk[d[e] * H + hd]));
        eb[t] = ex;
        atomicAdd(&den[d[e] * H + hd], ex);
    }
}
// pass 3: agg[dst,h,c] += h[src,h,c] * ex / (den[dst,h]+1e-16)
__global__ void k_edge_agg(const int* __restrict__ s, const int* __restrict__ d,
                           const float* __restrict__ eb, const float* __restrict__ den,
                           const float* __restrict__ h, float* __restrict__ agg, int H) {
    int HC = H * HIDC;
    int total = ETOT * HC;
    for (int t = gtid(); t < total; t += gstride()) {
        int e = t / HC, j = t - e * HC;
        int hd = j >> 7;
        int ds = d[e];
        float w = eb[e * H + hd] / (den[ds * H + hd] + 1e-16f);
        atomicAdd(&agg[(size_t)ds * HC + j], h[(size_t)s[e] * HC + j] * w);
    }
}
// epilogues: concat (out = agg + b) / head-mean (out = mean_h agg + b)
__global__ void k_concat_ep(const float* __restrict__ agg, const float* __restrict__ b,
                            float* __restrict__ out, int D) {
    int total = NN * D;
    for (int t = gtid(); t < total; t += gstride()) {
        int j = t % D;
        out[t] = agg[t] + b[j];
    }
}
__global__ void k_mean_ep(const float* __restrict__ agg, const float* __restrict__ b,
                          float* __restrict__ out, int H) {
    int total = NN * HIDC;
    float inv = 1.0f / (float)H;
    for (int t = gtid(); t < total; t += gstride()) {
        int n = t >> 7, c = t & (HIDC - 1);
        float s = 0.f;
        for (int hd = 0; hd < H; ++hd) s += agg[(size_t)n * H * HIDC + hd * HIDC + c];
        out[t] = s * inv + b[c];
    }
}

// ------------------------------- batchnorm ----------------------------------
__global__ void k_bn_stats(const float* __restrict__ x, float* __restrict__ mu,
                           float* __restrict__ rinv, int D) {
    __shared__ float s1[256], s2[256];
    int d = blockIdx.x;              // one block per column
    float sum = 0.f, ss = 0.f;
    for (int r = threadIdx.x; r < NN; r += blockDim.x) {
        float v = x[(size_t)r * D + d];
        sum += v; ss += v * v;
    }
    s1[threadIdx.x] = sum; s2[threadIdx.x] = ss;
    __syncthreads();
    for (int o = 128; o > 0; o >>= 1) {
        if (threadIdx.x < o) { s1[threadIdx.x] += s1[threadIdx.x + o];
                               s2[threadIdx.x] += s2[threadIdx.x + o]; }
        __syncthreads();
    }
    if (threadIdx.x == 0) {
        float m = s1[0] / (float)NN;
        float v = s2[0] / (float)NN - m * m;
        mu[d] = m;
        rinv[d] = rsqrtf(v + 1e-5f);
    }
}
__global__ void k_bn_apply(const float* __restrict__ x, const float* __restrict__ mu,
                           const float* __restrict__ rinv, const float* __restrict__ g,
                           const float* __restrict__ bt, float* __restrict__ y, int D) {
    int total = NN * D;
    for (int t = gtid(); t < total; t += gstride()) {
        int j = t % D;
        float v = (x[t] - mu[j]) * rinv[j] * g[j] + bt[j];
        y[t] = fmaxf(v, 0.0f);
    }
}

__global__ void k_addrelu(const float* __restrict__ a, const float* __restrict__ b,
                          float* __restrict__ y, int n) {
    for (int t = gtid(); t < n; t += gstride()) y[t] = fmaxf(a[t] + b[t], 0.0f);
}

// edge feature gather: ef16[e, 0:128]=xf[src], [128:256]=xf[dst]
__global__ void k_edgefeat(const int* __restrict__ ei, const float* __restrict__ xf,
                           _Float16* __restrict__ ef) {
    int total = EE * 256;
    for (int t = gtid(); t < total; t += gstride()) {
        int e = t >> 8, j = t & 255;
        int node = (j < HIDC) ? ei[e] : ei[EE + e];
        ef[t] = (_Float16)xf[(size_t)node * HIDC + (j & (HIDC - 1))];
    }
}

// scatter-mean pooling
__global__ void k_pool(const float* __restrict__ xf, const int* __restrict__ batch,
                       float* __restrict__ sums, float* __restrict__ cnt) {
    int total = NN * HIDC;
    for (int t = gtid(); t < total; t += gstride()) {
        int n = t >> 7, c = t & (HIDC - 1);
        int b = batch[n];
        atomicAdd(&sums[b * HIDC + c], xf[t]);
        if (c == 0) atomicAdd(&cnt[b], 1.0f);
    }
}
__global__ void k_gdiv(const float* __restrict__ sums, const float* __restrict__ cnt,
                       float* __restrict__ g) {
    int total = BB * HIDC;
    for (int t = gtid(); t < total; t += gstride()) {
        int b = t >> 7;
        g[t] = sums[t] / fmaxf(cnt[b], 1.0f);
    }
}

// ------------------------------- host side ----------------------------------

static inline int nblk(long long n) {
    long long b = (n + 255) / 256;
    if (b > 1048576) b = 1048576;
    if (b < 1) b = 1;
    return (int)b;
}

extern "C" void kernel_launch(void* const* d_in, const int* in_sizes, int n_in,
                              void* d_out, int out_size, void* d_ws, size_t ws_size,
                              hipStream_t stream) {
    (void)in_sizes; (void)n_in; (void)out_size; (void)ws_size;
    const float* xin  = (const float*)d_in[0];
    const int*   ei   = (const int*)d_in[1];
    const int*   batch= (const int*)d_in[3];
    #define P(i) ((const float*)d_in[i])

    // workspace carve-out (256B aligned)
    char* wp = (char*)d_ws;
    auto alloc = [&](size_t bytes) -> void* {
        void* r = (void*)wp; wp += (bytes + 255) & ~(size_t)255; return r;
    };
    _Float16* feat16 = (_Float16*)alloc((size_t)NN * 128 * 2);
    _Float16* wt16   = (_Float16*)alloc((size_t)512 * 512 * 2);
    float*    hbuf   = (float*)alloc((size_t)NN * 512 * 4);
    float*    agg    = (float*)alloc((size_t)NN * 512 * 4);
    float*    aS     = (float*)alloc((size_t)NN * 4 * 4);
    float*    aD     = (float*)alloc((size_t)NN * 4 * 4);
    int*      srcA   = (int*)alloc((size_t)ETOT * 4);
    int*      dstA   = (int*)alloc((size_t)ETOT * 4);
    float*    ebuf   = (float*)alloc((size_t)ETOT * 4 * 4);
    unsigned* mkey   = (unsigned*)alloc((size_t)NN * 4 * 4);
    float*    den    = (float*)alloc((size_t)NN * 4 * 4);
    float*    x1     = (float*)alloc((size_t)NN * 512 * 4);
    _Float16* x1h    = (_Float16*)alloc((size_t)NN * 512 * 2);
    float*    x2     = (float*)alloc((size_t)NN * 128 * 4);
    _Float16* x2h    = (_Float16*)alloc((size_t)NN * 128 * 2);
    float*    x3     = (float*)alloc((size_t)NN * 128 * 4);
    float*    rs     = (float*)alloc((size_t)NN * 128 * 4);
    _Float16* xfh    = (_Float16*)alloc((size_t)NN * 128 * 2);
    float*    mu     = (float*)alloc(512 * 4);
    float*    rinv   = (float*)alloc(512 * 4);
    float*    tmp    = (float*)alloc((size_t)NN * 128 * 4);
    _Float16* tmph   = (_Float16*)alloc((size_t)NN * 128 * 2);
    _Float16* efh    = (_Float16*)alloc((size_t)EE * 256 * 2);
    float*    ehid   = (float*)alloc((size_t)EE * 128 * 4);
    _Float16* ehidh  = (_Float16*)alloc((size_t)EE * 128 * 2);
    float*    psums  = (float*)alloc((size_t)BB * 128 * 4);
    float*    pcnt   = (float*)alloc((size_t)BB * 4);
    float*    gemb   = (float*)alloc((size_t)BB * 128 * 4);
    _Float16* gembh  = (_Float16*)alloc((size_t)BB * 128 * 2);
    float*    sph    = (float*)alloc((size_t)BB * 128 * 4);
    _Float16* sphh   = (_Float16*)alloc((size_t)BB * 128 * 2);

    // output slices
    float* out   = (float*)d_out;
    float* o_td  = out;
    float* o_pc  = out + (size_t)NN * 8;
    float* o_nv  = out + (size_t)NN * 16;
    float* o_ps  = out + (size_t)NN * 27;
    float* o_xf  = out + (size_t)NN * 28;
    float* o_etp = out + (size_t)NN * 156;
    float* o_etc = out + (size_t)NN * 156 + (size_t)EE * 3;
    float* o_sp  = out + (size_t)NN * 156 + (size_t)EE * 8;

    auto gemmW = [&](const float* W, const float* bias, const _Float16* Ah, float* Cout,
                     int M, int K, int Nreal, int Npad, int ep) {
        k_wprep<<<nblk((long long)Npad * K), 256, 0, stream>>>(W, wt16, K, Nreal, Npad);
        int waves = ((M + 15) / 16) * (Npad / 16);
        k_gemm<<<dim3((waves + 7) / 8), dim3(256), 0, stream>>>(
            Ah, wt16, bias, Cout, M, K, Npad, Nreal, ep);
    };

    auto gatLayer = [&](const _Float16* Xh, int Kin, int H, const float* W,
                        const float* atS, const float* atD, const float* b,
                        bool concat, float* outPre) {
        int HC = H * HIDC;
        gemmW(W, nullptr, Xh, hbuf, NN, Kin, HC, HC, 0);
        k_att_nodes<<<nblk((long long)NN * H), 256, 0, stream>>>(hbuf, atS, atD, aS, aD, H);
        k_fill_u32<<<nblk(NN * 4), 256, 0, stream>>>(mkey, NEG_INF_KEY, NN * 4);
        k_fill_f32<<<nblk(NN * 4), 256, 0, stream>>>(den, 0.0f, NN * 4);
        k_fill_f32<<<nblk((long long)NN * HC), 256, 0, stream>>>(agg, 0.0f, NN * HC);
        k_edge_max<<<nblk((long long)ETOT * H), 256, 0, stream>>>(srcA, dstA, aS, aD, ebuf, mkey, H);
        k_edge_exp<<<nblk((long long)ETOT * H), 256, 0, stream>>>(dstA, ebuf, mkey, den, H);
        k_edge_agg<<<nblk((long long)ETOT * HC), 256, 0, stream>>>(srcA, dstA, ebuf, den, hbuf, agg, H);
        if (concat) k_concat_ep<<<nblk((long long)NN * HC), 256, 0, stream>>>(agg, b, outPre, HC);
        else        k_mean_ep<<<nblk((long long)NN * HIDC), 256, 0, stream>>>(agg, b, outPre, H);
    };

    auto bnRelu = [&](const float* xpre, const float* g, const float* bt, float* y, int D) {
        k_bn_stats<<<dim3(D), dim3(256), 0, stream>>>(xpre, mu, rinv, D);
        k_bn_apply<<<nblk((long long)NN * D), 256, 0, stream>>>(xpre, mu, rinv, g, bt, y, D);
    };

    // ---- featurize + edge list with self loops ----
    k_build_edges<<<nblk(ETOT), 256, 0, stream>>>(ei, srcA, dstA);
    k_feat<<<nblk(NN), 256, 0, stream>>>(xin, P(4), P(5), P(6), P(7), P(8), feat16);

    // ---- GAT1: 128 -> 4x128 concat -> BN -> ReLU ----
    gatLayer(feat16, 128, 4, P(9), P(10), P(11), P(12), true, hbuf /* reuse as pre-bn */);
    bnRelu(hbuf, P(21), P(22), x1, 512);
    k_f32_to_f16<<<nblk((long long)NN * 512), 256, 0, stream>>>(x1, x1h, NN * 512);

    // ---- GAT2: 512 -> 2x128 mean -> BN -> ReLU ----
    gatLayer(x1h, 512, 2, P(13), P(14), P(15), P(16), false, tmp);
    bnRelu(tmp, P(23), P(24), x2, 128);
    k_f32_to_f16<<<nblk((long long)NN * 128), 256, 0, stream>>>(x2, x2h, NN * 128);

    // ---- GAT3: 128 -> 1x128 concat -> BN -> ReLU ----
    gatLayer(x2h, 128, 1, P(17), P(18), P(19), P(20), true, tmp);
    bnRelu(tmp, P(25), P(26), x3, 128);

    // ---- xf = relu(x3 + rescale(x1)) ----
    gemmW(P(27), P(28), x1h, rs, NN, 512, 128, 128, 0);
    k_addrelu<<<nblk((long long)NN * 128), 256, 0, stream>>>(x3, rs, o_xf, NN * 128);
    k_f32_to_f16<<<nblk((long long)NN * 128), 256, 0, stream>>>(o_xf, xfh, NN * 128);

    // ---- node heads ----
    gemmW(P(29), P(30), xfh, tmp, NN, 128, 128, 128, 1);
    k_f32_to_f16<<<nblk((long long)NN * 128), 256, 0, stream>>>(tmp, tmph, NN * 128);
    gemmW(P(31), P(32), tmph, o_td, NN, 128, 8, 16, 0);

    gemmW(P(33), P(34), xfh, tmp, NN, 128, 128, 128, 1);
    k_f32_to_f16<<<nblk((long long)NN * 128), 256, 0, stream>>>(tmp, tmph, NN * 128);
    gemmW(P(35), P(36), tmph, o_pc, NN, 128, 8, 16, 0);

    gemmW(P(37), P(38), xfh, tmp, NN, 128, 128, 128, 1);
    k_f32_to_f16<<<nblk((long long)NN * 128), 256, 0, stream>>>(tmp, tmph, NN * 128);
    gemmW(P(39), P(40), tmph, o_nv, NN, 128, 11, 16, 0);

    gemmW(P(41), P(42), xfh, tmp, NN, 128, 64, 64, 1);
    k_f32_to_f16<<<nblk((long long)NN * 64), 256, 0, stream>>>(tmp, tmph, NN * 64);
    gemmW(P(43), P(44), tmph, o_ps, NN, 64, 1, 16, 2);   // sigmoid

    // ---- edge heads ----
    k_edgefeat<<<nblk((long long)EE * 256), 256, 0, stream>>>(ei, o_xf, efh);
    gemmW(P(49), P(50), efh, ehid, EE, 256, 128, 128, 1);
    k_f32_to_f16<<<nblk((long long)EE * 128), 256, 0, stream>>>(ehid, ehidh, EE * 128);
    gemmW(P(51), P(52), ehidh, o_etp, EE, 128, 3, 16, 0);

    gemmW(P(45), P(46), efh, ehid, EE, 256, 128, 128, 1);
    k_f32_to_f16<<<nblk((long long)EE * 128), 256, 0, stream>>>(ehid, ehidh, EE * 128);
    gemmW(P(47), P(48), ehidh, o_etc, EE, 128, 5, 16, 0);

    // ---- graph pooling + shape params ----
    k_fill_f32<<<nblk(BB * 128), 256, 0, stream>>>(psums, 0.0f, BB * 128);
    k_fill_f32<<<nblk(BB), 256, 0, stream>>>(pcnt, 0.0f, BB);
    k_pool<<<nblk((long long)NN * 128), 256, 0, stream>>>(o_xf, batch, psums, pcnt);
    k_gdiv<<<nblk(BB * 128), 256, 0, stream>>>(psums, pcnt, gemb);
    k_f32_to_f16<<<nblk(BB * 128), 256, 0, stream>>>(gemb, gembh, BB * 128);
    gemmW(P(53), P(54), gembh, sph, BB, 128, 128, 128, 1);
    k_f32_to_f16<<<nblk(BB * 128), 256, 0, stream>>>(sph, sphh, BB * 128);
    gemmW(P(55), P(56), sphh, o_sp, BB, 128, 4, 16, 0);

    #undef P
}
